// BasicTFBlock_61701500174515
// MI455X (gfx1250) — compile-verified, matrix operand order checked by reading, not run on hardware
//
#include <hip/hip_runtime.h>
#include <hip/hip_bf16.h>
#include <math.h>

#define WSZ 2
#define NHD 4
#define CI  48
#define CIP 64                 // padded K for conv1/res
#define CC  96
#define BBATCH 2
#define HSZ 40
#define SPV (HSZ*HSZ*HSZ)      // 64000 voxels per batch
#define VTOT (BBATCH*SPV)      // 128000 tokens
#define MT (VTOT/16)           // 8000 M-tiles
#define HD  24                 // head dim
#define NWIN (HSZ/WSZ)         // 20 windows per axis

typedef __attribute__((ext_vector_type(16))) __bf16          v16bf;
typedef __attribute__((ext_vector_type(8)))  float           v8f;
typedef __attribute__((ext_vector_type(8)))  unsigned short  v8us;
typedef __attribute__((ext_vector_type(16))) unsigned short  v16us;
typedef __attribute__((ext_vector_type(4)))  unsigned int    v4u;
typedef __attribute__((ext_vector_type(8)))  unsigned int    v8u;
typedef __attribute__((ext_vector_type(4)))  int             v4i;
typedef __attribute__((ext_vector_type(8)))  int             v8i;

__device__ __forceinline__ __bf16 f2bf(float f) {
  unsigned u = __builtin_bit_cast(unsigned, f);
  unsigned r = (u + 0x7FFFu + ((u >> 16) & 1u)) >> 16;
  return __builtin_bit_cast(__bf16, (unsigned short)r);
}
// pack two f32 into two rounded bf16 in one dword (lo, hi)
__device__ __forceinline__ unsigned pk2bf(float a, float b) {
  unsigned ua = __builtin_bit_cast(unsigned, a);
  unsigned ub = __builtin_bit_cast(unsigned, b);
  unsigned ra = (ua + 0x7FFFu + ((ua >> 16) & 1u)) >> 16;
  unsigned rb = (ub + 0x7FFFu + ((ub >> 16) & 1u)) & 0xFFFF0000u;
  return (ra & 0xFFFFu) | rb;
}
// unpack dword holding two bf16 -> two f32
__device__ __forceinline__ void up2(unsigned u, float& a, float& b) {
  a = __builtin_bit_cast(float, u << 16);
  b = __builtin_bit_cast(float, u & 0xFFFF0000u);
}

// A fragment from row-major bf16 row (two b128 loads). kb = (lane>>4)*8.
__device__ __forceinline__ v16bf load_a16(const __bf16* __restrict__ row, int kb) {
  v8us lo = *(const v8us*)(row + kb);
  v8us hi = *(const v8us*)(row + kb + 16);
  return __builtin_bit_cast(v16bf,
      __builtin_shufflevector(lo, hi, 0,1,2,3,4,5,6,7,8,9,10,11,12,13,14,15));
}
// fragment stored contiguously (32B per lane) -> two b128 loads
__device__ __forceinline__ v16bf load_pack16(const __bf16* __restrict__ p) {
  v8us lo = *(const v8us*)p;
  v8us hi = *(const v8us*)(p + 8);
  return __builtin_bit_cast(v16bf,
      __builtin_shufflevector(lo, hi, 0,1,2,3,4,5,6,7,8,9,10,11,12,13,14,15));
}

#define WMMA_BF16(a, b, c) \
  __builtin_amdgcn_wmma_f32_16x16x32_bf16(false, (a), false, (b), (short)0, (c), false, false)

// ---------------- Tensor Data Mover: async DMA of a 2D tile into LDS --------
// 5-arg builtin on ROCm 7.2 (clang-22); 6-arg on therock/clang-23 (detected
// via its TDM header). Groups 2/3 are NULL-equivalent (zero) for 2D tensors.
__device__ __forceinline__ void tdm_load_to_lds(v4u g0, v8u g1) {
  v8i g1i = __builtin_bit_cast(v8i, g1);
  v4i z4 = {};
#if __has_include(<hip/amd_detail/amd_gfx1250_TDM.h>)
  v8i z8 = {};
  __builtin_amdgcn_tensor_load_to_lds(g0, g1i, z4, z4, z8, 0);
#else
  __builtin_amdgcn_tensor_load_to_lds(g0, g1i, z4, z4, 0);
#endif
}

// ---------------------------------------------------------------- pack / fold
__global__ void bn_fold_kernel(const float* __restrict__ bn, const float* __restrict__ bias,
                               float* __restrict__ scale, float* __restrict__ shift) {
  int c = threadIdx.x;
  if (c >= CC) return;
  float g = bn[c], b = bn[CC + c], m = bn[2 * CC + c], v = bn[3 * CC + c];
  float s = g * rsqrtf(v + 1e-5f);
  scale[c] = s;
  shift[c] = b - m * s + bias[c] * s;
}

// x (B,CI,spatial) f32 -> token-major bf16 with K padded to 64 (zeros)
__global__ void pack_x_kernel(const float* __restrict__ x, __bf16* __restrict__ xtok) {
  int total = VTOT * CIP;
  for (int idx = blockIdx.x * blockDim.x + threadIdx.x; idx < total;
       idx += gridDim.x * blockDim.x) {
    int c = idx & (CIP - 1);
    int v = idx >> 6;
    int b = v / SPV, sp = v % SPV;
    float val = (c < CI) ? x[((size_t)b * CI + c) * SPV + sp] : 0.0f;
    xtok[idx] = f2bf(val);
  }
}

// conv weight (O=96, I, taps) -> WMMA fragment layout [tap][kc][nt][lane][16]
__global__ void pack_conv_frag_kernel(const float* __restrict__ w, __bf16* __restrict__ out,
                                      int I, int KC, int taps) {
  int total = taps * KC * 6 * 512;
  for (int idx = blockIdx.x * blockDim.x + threadIdx.x; idx < total;
       idx += gridDim.x * blockDim.x) {
    int i    = idx & 15;
    int lane = (idx >> 4) & 31;
    int tile = idx >> 9;
    int nt   = tile % 6;
    int kc   = (tile / 6) % KC;
    int tap  = tile / (6 * KC);
    int k = kc * 32 + ((lane >> 4) << 3) + i + ((i >= 8) ? 8 : 0);
    int o = nt * 16 + (lane & 15);
    float v = (k < I) ? w[((size_t)o * I + k) * taps + tap] : 0.0f;
    out[idx] = f2bf(v);
  }
}

// linear weight (N,K) -> fragment layout [kc][nt][lane][16]
__global__ void pack_lin_frag_kernel(const float* __restrict__ w, __bf16* __restrict__ out,
                                     int N, int K) {
  int Nt = N >> 4, KC = K >> 5;
  int total = KC * Nt * 512;
  for (int idx = blockIdx.x * blockDim.x + threadIdx.x; idx < total;
       idx += gridDim.x * blockDim.x) {
    int i    = idx & 15;
    int lane = (idx >> 4) & 31;
    int tile = idx >> 9;
    int nt   = tile % Nt;
    int kc   = tile / Nt;
    int k = kc * 32 + ((lane >> 4) << 3) + i + ((i >= 8) ? 8 : 0);
    int n = nt * 16 + (lane & 15);
    out[idx] = f2bf(w[(size_t)n * K + k]);
  }
}

// ------------------------------------------------------------ residual 1x1x1
// wave per M-tile, 6 N-tiles per wave (full C=96), K = 64 (padded)
__global__ void conv_res_kernel(const __bf16* __restrict__ xtok, const __bf16* __restrict__ w,
                                const float* __restrict__ scale, const float* __restrict__ shift,
                                float* __restrict__ res_tok) {
  int mt = blockIdx.x * (blockDim.x >> 5) + (threadIdx.x >> 5);
  int lane = threadIdx.x & 31;
  if (mt >= MT) return;
  int m = lane & 15, kb = (lane >> 4) << 3;
  const __bf16* ar = xtok + (size_t)(mt * 16 + m) * CIP;
  const v8f vz = {};
  v8f acc[6] = {vz, vz, vz, vz, vz, vz};
#pragma unroll
  for (int kc = 0; kc < 2; ++kc) {
    v16bf a = load_a16(ar + kc * 32, kb);
    const __bf16* wt = w + (size_t)kc * 6 * 512 + lane * 16;
#pragma unroll
    for (int ntl = 0; ntl < 6; ++ntl) {
      v16bf bfrag = load_pack16(wt + ntl * 512);
      acc[ntl] = WMMA_BF16(a, bfrag, acc[ntl]);
    }
  }
  int n = lane & 15, mb = (lane >> 4) << 3;
#pragma unroll
  for (int ntl = 0; ntl < 6; ++ntl) {
    int col = ntl * 16 + n;
    float sc = scale[col], sh = shift[col];
#pragma unroll
    for (int r = 0; r < 8; ++r) {
      int row = mt * 16 + mb + r;
      res_tok[(size_t)row * CC + col] = fmaxf(acc[ntl][r] * sc + sh, 0.0f);
    }
  }
}

// ------------------------------------------------------------- conv1 3x3x3
__global__ void conv1_kernel(const __bf16* __restrict__ xtok, const __bf16* __restrict__ w,
                             const float* __restrict__ scale, const float* __restrict__ shift,
                             float* __restrict__ cx, float* __restrict__ t) {
  int mt = blockIdx.x * (blockDim.x >> 5) + (threadIdx.x >> 5);
  int lane = threadIdx.x & 31;
  if (mt >= MT) return;
  int m = lane & 15, kb = (lane >> 4) << 3;
  int voxel = mt * 16 + m;
  int b = voxel / SPV, sp = voxel % SPV;
  int hc = sp / (HSZ * HSZ), wc = (sp / HSZ) % HSZ, tc = sp % HSZ;
  v16us zu = {};
  const v16bf azero = __builtin_bit_cast(v16bf, zu);
  const v8f vz = {};
  v8f acc[6] = {vz, vz, vz, vz, vz, vz};
  for (int tap = 0; tap < 27; ++tap) {
    int dd = tap / 9 - 1, dh = (tap / 3) % 3 - 1, dw = tap % 3 - 1;
    int hh = hc + dd, ww = wc + dh, tt = tc + dw;
    bool valid = (unsigned)hh < HSZ && (unsigned)ww < HSZ && (unsigned)tt < HSZ;
    const __bf16* ar = xtok + (size_t)(b * SPV + (hh * HSZ + ww) * HSZ + tt) * CIP;
    const __bf16* wt = w + (size_t)tap * 2 * 6 * 512 + lane * 16;
    __builtin_prefetch(wt, 0, 0);
#pragma unroll
    for (int kc = 0; kc < 2; ++kc) {
      v16bf a = valid ? load_a16(ar + kc * 32, kb) : azero;
      const __bf16* wk = wt + (size_t)kc * 6 * 512;
#pragma unroll
      for (int ntl = 0; ntl < 6; ++ntl) {
        v16bf bfrag = load_pack16(wk + ntl * 512);
        acc[ntl] = WMMA_BF16(a, bfrag, acc[ntl]);
      }
    }
  }
  int n = lane & 15, mb = (lane >> 4) << 3;
#pragma unroll
  for (int ntl = 0; ntl < 6; ++ntl) {
    int col = ntl * 16 + n;
    float sc = scale[col], sh = shift[col];
#pragma unroll
    for (int r = 0; r < 8; ++r) {
      int row = mt * 16 + mb + r;
      float v = fmaxf(acc[ntl][r] * sc + sh, 0.0f);
      cx[(size_t)row * CC + col] = v;
      t[(size_t)row * CC + col]  = v;
    }
  }
}

// ------------------------------------- conv2 3x3x3 (+res, NCDHW output)
__global__ void conv2_kernel(const __bf16* __restrict__ src, const __bf16* __restrict__ w,
                             const float* __restrict__ scale, const float* __restrict__ shift,
                             const float* __restrict__ res_tok, float* __restrict__ out) {
  int mt = blockIdx.x * (blockDim.x >> 5) + (threadIdx.x >> 5);
  int lane = threadIdx.x & 31;
  if (mt >= MT) return;
  int m = lane & 15, kb = (lane >> 4) << 3;
  int voxel = mt * 16 + m;
  int b = voxel / SPV, sp = voxel % SPV;
  int hc = sp / (HSZ * HSZ), wc = (sp / HSZ) % HSZ, tc = sp % HSZ;
  v16us zu = {};
  const v16bf azero = __builtin_bit_cast(v16bf, zu);
  const v8f vz = {};
  v8f acc[6] = {vz, vz, vz, vz, vz, vz};
  for (int tap = 0; tap < 27; ++tap) {
    int dd = tap / 9 - 1, dh = (tap / 3) % 3 - 1, dw = tap % 3 - 1;
    int hh = hc + dd, ww = wc + dh, tt = tc + dw;
    bool valid = (unsigned)hh < HSZ && (unsigned)ww < HSZ && (unsigned)tt < HSZ;
    const __bf16* ar = src + (size_t)(b * SPV + (hh * HSZ + ww) * HSZ + tt) * CC;
    const __bf16* wt = w + (size_t)tap * 3 * 6 * 512 + lane * 16;
    __builtin_prefetch(wt, 0, 0);
#pragma unroll
    for (int kc = 0; kc < 3; ++kc) {
      v16bf a = valid ? load_a16(ar + kc * 32, kb) : azero;
      const __bf16* wk = wt + (size_t)kc * 6 * 512;
#pragma unroll
      for (int ntl = 0; ntl < 6; ++ntl) {
        v16bf bfrag = load_pack16(wk + ntl * 512);
        acc[ntl] = WMMA_BF16(a, bfrag, acc[ntl]);
      }
    }
  }
  int n = lane & 15, mb = (lane >> 4) << 3;
#pragma unroll
  for (int ntl = 0; ntl < 6; ++ntl) {
    int col = ntl * 16 + n;
    float sc = scale[col], sh = shift[col];
#pragma unroll
    for (int r = 0; r < 8; ++r) {
      int row = mt * 16 + mb + r;
      int rb = row / SPV, rsp = row % SPV;
      float v = fmaxf(acc[ntl][r] * sc + sh, 0.0f) + res_tok[(size_t)row * CC + col];
      out[((size_t)rb * CC + col) * SPV + rsp] = v;
    }
  }
}

// --------------------------------------------------------------- LayerNorm
__global__ void ln_kernel(const float* __restrict__ t, const float* __restrict__ g,
                          const float* __restrict__ bparm, __bf16* __restrict__ out) {
  int v = blockIdx.x * blockDim.x + threadIdx.x;
  if (v >= VTOT) return;
  const float4* r4 = (const float4*)(t + (size_t)v * CC);
  float s = 0.0f, s2 = 0.0f;
#pragma unroll 6
  for (int i = 0; i < CC / 4; ++i) {
    float4 q = r4[i];
    s  += q.x + q.y + q.z + q.w;
    s2 += q.x * q.x + q.y * q.y + q.z * q.z + q.w * q.w;
  }
  float mu  = s * (1.0f / CC);
  float var = s2 * (1.0f / CC) - mu * mu;
  float inv = rsqrtf(var + 1e-5f);
  const float4* g4 = (const float4*)g;
  const float4* b4 = (const float4*)bparm;
  uint2* o = (uint2*)(out + (size_t)v * CC);
#pragma unroll 6
  for (int i = 0; i < CC / 4; ++i) {
    float4 q = r4[i], gg = g4[i], bb = b4[i];
    float y0 = (q.x - mu) * inv * gg.x + bb.x;
    float y1 = (q.y - mu) * inv * gg.y + bb.y;
    float y2 = (q.z - mu) * inv * gg.z + bb.z;
    float y3 = (q.w - mu) * inv * gg.w + bb.w;
    uint2 u;
    u.x = pk2bf(y0, y1);
    u.y = pk2bf(y2, y3);
    o[i] = u;
  }
}

// ------------------------------------------------- generic WMMA linear layer
// Block = 4 waves (4 M-tiles), one group of 6 N-tiles; packed B DMA'd into
// LDS by the Tensor Data Mover (2D tile: KC rows x 3072 bf16, row stride
// Nt*512 elements), completed with s_wait_tensorcnt before the barrier.
// mode 0: obf = A@B + bias (first 96 cols * qscale)   [qkv]
// mode 1: tacc += A@B + bias                          [proj, fc2]
// mode 2: obf = gelu(A@B + bias)                      [fc1]
__global__ void linear_kernel(const __bf16* __restrict__ A, const __bf16* __restrict__ Bw,
                              const float* __restrict__ bias, int N, int K,
                              float* __restrict__ tacc, __bf16* __restrict__ obf,
                              int mode, float qscale) {
  extern __shared__ char smem_raw[];
  __bf16* sB = (__bf16*)smem_raw;
  int tid = threadIdx.x;
  int wid = tid >> 5, lane = tid & 31;
  int Nt = N >> 4, KC = K >> 5;
  int ngroups = Nt / 6;
  int g = blockIdx.x % ngroups;
  int mt = (blockIdx.x / ngroups) * 4 + wid;

  if (wid == 0) {
    unsigned ldsOff = (unsigned)(size_t)(void*)sB;
    size_t ga = (size_t)(const void*)(Bw + (size_t)6 * g * 512);
    v4u g0;
    g0[0] = 1u;                                                // count=1, no gather
    g0[1] = ldsOff;                                            // lds_addr (bytes)
    g0[2] = (unsigned)ga;                                      // global_addr[31:0]
    g0[3] = (unsigned)((ga >> 32) & 0x01FFFFFFu) | (2u << 30); // ga[56:32] | type=2
    v8u g1;
    g1[0] = 1u << 16;                    // data_size = 2 bytes
    g1[1] = (3072u & 0xFFFFu) << 16;     // tensor_dim0[15:0] (3072 elems/row)
    g1[2] = ((unsigned)KC) << 16;        // tensor_dim0[31:16]=0 | tensor_dim1 = KC
    g1[3] = 3072u << 16;                 // tensor_dim1[31:16]=0 | tile_dim0 = 3072
    g1[4] = (unsigned)KC;                // tile_dim1 = KC | tile_dim2 = 0
    g1[5] = (unsigned)(Nt * 512);        // tensor_dim0_stride (elements)
    g1[6] = 0u;
    g1[7] = 0u;
    tdm_load_to_lds(g0, g1);
    __builtin_amdgcn_s_wait_tensorcnt(0);
  }
  __syncthreads();

  int m = lane & 15, kb = (lane >> 4) << 3;
  const __bf16* Am = A + (size_t)(mt * 16 + m) * K;
  __builtin_prefetch(Am, 0, 0);
  const v8f vz = {};
  v8f acc[6] = {vz, vz, vz, vz, vz, vz};
  for (int kc = 0; kc < KC; ++kc) {
    v16bf a = load_a16(Am + kc * 32, kb);
    const __bf16* bp = sB + (size_t)kc * 3072 + lane * 16;
#pragma unroll
    for (int ntl = 0; ntl < 6; ++ntl) {
      v16bf bfrag = load_pack16(bp + ntl * 512);
      acc[ntl] = WMMA_BF16(a, bfrag, acc[ntl]);
    }
  }

  int n = lane & 15, mb = (lane >> 4) << 3;
#pragma unroll
  for (int ntl = 0; ntl < 6; ++ntl) {
    int col = g * 96 + ntl * 16 + n;
    float bi = bias[col];
#pragma unroll
    for (int r = 0; r < 8; ++r) {
      int row = mt * 16 + mb + r;
      float v = acc[ntl][r] + bi;
      if (mode == 0) {
        if (col < CC) v *= qscale;
        obf[(size_t)row * N + col] = f2bf(v);
      } else if (mode == 1) {
        tacc[(size_t)row * N + col] += v;
      } else {
        v = 0.5f * v * (1.0f + erff(v * 0.70710678118654752f));
        obf[(size_t)row * N + col] = f2bf(v);
      }
    }
  }
}

// ---------------------------------------------------- window attention (N=8)
__global__ void attn_kernel(const __bf16* __restrict__ qkv, const float* __restrict__ rpb,
                            __bf16* __restrict__ aout, int shift) {
  int idx = blockIdx.x * blockDim.x + threadIdx.x;
  const int nwin = BBATCH * NWIN * NWIN * NWIN;   // 16000
  if (idx >= nwin * NHD * 8) return;
  int n    = idx & 7;
  int head = (idx >> 3) & 3;
  int w    = idx >> 5;
  int b    = w / (NWIN * NWIN * NWIN);
  int wr   = w % (NWIN * NWIN * NWIN);
  int wh = wr / (NWIN * NWIN), ww = (wr / NWIN) % NWIN, wt = wr % NWIN;

  auto vox = [&](int j, int& ph, int& pw, int& pt) -> int {
    int dh = j >> 2, dw = (j >> 1) & 1, dt = j & 1;
    ph = wh * WSZ + dh; pw = ww * WSZ + dw; pt = wt * WSZ + dt;
    int oh = ph + shift; if (oh >= HSZ) oh -= HSZ;
    int ow = pw + shift; if (ow >= HSZ) ow -= HSZ;
    int ot = pt + shift; if (ot >= HSZ) ot -= HSZ;
    return b * SPV + (oh * HSZ + ow) * HSZ + ot;
  };
  auto cat = [](int p) { return (p < HSZ - WSZ) ? 0 : ((p < HSZ - 1) ? 1 : 2); };

  int phn, pwn, ptn;
  int vn = vox(n, phn, pwn, ptn);
  int lbl_n = cat(phn) * 9 + cat(pwn) * 3 + cat(ptn);

  float q[HD];
  {
    const uint4* p = (const uint4*)(qkv + (size_t)vn * (3 * CC) + head * HD);
#pragma unroll
    for (int j = 0; j < 3; ++j) {
      uint4 u = p[j];
      up2(u.x, q[8 * j + 0], q[8 * j + 1]);
      up2(u.y, q[8 * j + 2], q[8 * j + 3]);
      up2(u.z, q[8 * j + 4], q[8 * j + 5]);
      up2(u.w, q[8 * j + 6], q[8 * j + 7]);
    }
  }
  int an = n >> 2, bn = (n >> 1) & 1, cn = n & 1;

  float a[8];
  float amax = -1e30f;
  for (int mm = 0; mm < 8; ++mm) {
    int phm, pwm, ptm;
    int vm = vox(mm, phm, pwm, ptm);
    const uint4* kp = (const uint4*)(qkv + (size_t)vm * (3 * CC) + CC + head * HD);
    float s = 0.0f;
#pragma unroll
    for (int j = 0; j < 3; ++j) {
      uint4 u = kp[j];
      float k0, k1;
      up2(u.x, k0, k1); s += q[8 * j + 0] * k0 + q[8 * j + 1] * k1;
      up2(u.y, k0, k1); s += q[8 * j + 2] * k0 + q[8 * j + 3] * k1;
      up2(u.z, k0, k1); s += q[8 * j + 4] * k0 + q[8 * j + 5] * k1;
      up2(u.w, k0, k1); s += q[8 * j + 6] * k0 + q[8 * j + 7] * k1;
    }
    int am = mm >> 2, bm = (mm >> 1) & 1, cm = mm & 1;
    int ridx = (an - am + 1) * 9 + (bn - bm + 1) * 3 + (cn - cm + 1);
    s += rpb[ridx * NHD + head];
    if (shift) {
      int lbl_m = cat(phm) * 9 + cat(pwm) * 3 + cat(ptm);
      if (lbl_m != lbl_n) s -= 100.0f;
    }
    a[mm] = s;
    amax = fmaxf(amax, s);
  }
  float ssum = 0.0f;
#pragma unroll
  for (int mm = 0; mm < 8; ++mm) { a[mm] = __expf(a[mm] - amax); ssum += a[mm]; }
  float inv = 1.0f / ssum;

  float o[HD];
#pragma unroll
  for (int d = 0; d < HD; ++d) o[d] = 0.0f;
  for (int mm = 0; mm < 8; ++mm) {
    int phm, pwm, ptm;
    int vm = vox(mm, phm, pwm, ptm);
    const uint4* vp = (const uint4*)(qkv + (size_t)vm * (3 * CC) + 2 * CC + head * HD);
    float aw = a[mm] * inv;
#pragma unroll
    for (int j = 0; j < 3; ++j) {
      uint4 u = vp[j];
      float v0, v1;
      up2(u.x, v0, v1); o[8 * j + 0] += aw * v0; o[8 * j + 1] += aw * v1;
      up2(u.y, v0, v1); o[8 * j + 2] += aw * v0; o[8 * j + 3] += aw * v1;
      up2(u.z, v0, v1); o[8 * j + 4] += aw * v0; o[8 * j + 5] += aw * v1;
      up2(u.w, v0, v1); o[8 * j + 6] += aw * v0; o[8 * j + 7] += aw * v1;
    }
  }
  uint4* op = (uint4*)(aout + (size_t)vn * CC + head * HD);
#pragma unroll
  for (int j = 0; j < 3; ++j) {
    uint4 u;
    u.x = pk2bf(o[8 * j + 0], o[8 * j + 1]);
    u.y = pk2bf(o[8 * j + 2], o[8 * j + 3]);
    u.z = pk2bf(o[8 * j + 4], o[8 * j + 5]);
    u.w = pk2bf(o[8 * j + 6], o[8 * j + 7]);
    op[j] = u;
  }
}

// ---------------------------------------------------------------- cx + t -> bf16
__global__ void cxsum_kernel(const float4* __restrict__ cx4, const float4* __restrict__ t4,
                             uint2* __restrict__ out) {
  int total = VTOT * CC / 4;
  for (int i = blockIdx.x * blockDim.x + threadIdx.x; i < total;
       i += gridDim.x * blockDim.x) {
    float4 a = cx4[i], b = t4[i];
    uint2 r;
    r.x = pk2bf(a.x + b.x, a.y + b.y);
    r.y = pk2bf(a.z + b.z, a.w + b.w);
    out[i] = r;
  }
}

// ================================================================= launcher
extern "C" void kernel_launch(void* const* d_in, const int* in_sizes, int n_in,
                              void* d_out, int out_size, void* d_ws, size_t ws_size,
                              hipStream_t stream) {
  (void)in_sizes; (void)n_in; (void)out_size; (void)ws_size;
  const float* x       = (const float*)d_in[0];
  const float* res_w   = (const float*)d_in[1];
  const float* res_b   = (const float*)d_in[2];
  const float* res_bn  = (const float*)d_in[3];
  const float* conv1_w = (const float*)d_in[4];
  const float* conv1_b = (const float*)d_in[5];
  const float* bn1     = (const float*)d_in[6];
  const float* conv2_w = (const float*)d_in[7];
  const float* conv2_b = (const float*)d_in[8];
  const float* bn2     = (const float*)d_in[9];
  const float* n1      = (const float*)d_in[10];
  const float* n2      = (const float*)d_in[11];
  const float* qkv_w   = (const float*)d_in[12];
  const float* qkv_b   = (const float*)d_in[13];
  const float* proj_w  = (const float*)d_in[14];
  const float* proj_b  = (const float*)d_in[15];
  const float* rpb     = (const float*)d_in[16];
  const float* fc1_w   = (const float*)d_in[17];
  const float* fc1_b   = (const float*)d_in[18];
  const float* fc2_w   = (const float*)d_in[19];
  const float* fc2_b   = (const float*)d_in[20];

  char* ws = (char*)d_ws;
  size_t off = 0;
  auto alloc = [&](size_t bytes) { size_t o = off; off += (bytes + 255) & ~(size_t)255; return o; };

  size_t o_xtok = alloc((size_t)VTOT * CIP * 2);
  size_t o_wres = alloc((size_t)2 * 6 * 512 * 2);
  size_t o_wc1  = alloc((size_t)27 * 2 * 6 * 512 * 2);
  size_t o_wc2  = alloc((size_t)27 * 3 * 6 * 512 * 2);
  size_t o_wqkv = alloc((size_t)2 * 3 * 18 * 512 * 2);
  size_t o_wprj = alloc((size_t)2 * 3 * 6 * 512 * 2);
  size_t o_wfc1 = alloc((size_t)2 * 3 * 24 * 512 * 2);
  size_t o_wfc2 = alloc((size_t)2 * 12 * 6 * 512 * 2);
  size_t o_scr  = alloc(CC * 4); size_t o_shr = alloc(CC * 4);
  size_t o_sc1  = alloc(CC * 4); size_t o_sh1 = alloc(CC * 4);
  size_t o_sc2  = alloc(CC * 4); size_t o_sh2 = alloc(CC * 4);
  size_t o_cx   = alloc((size_t)VTOT * CC * 4);
  size_t o_t    = alloc((size_t)VTOT * CC * 4);
  size_t o_res  = alloc((size_t)VTOT * CC * 4);
  size_t o_hbf  = alloc((size_t)VTOT * CC * 2);
  size_t o_qkv  = alloc((size_t)VTOT * 288 * 2);
  size_t o_aob  = alloc((size_t)VTOT * CC * 2);
  size_t o_hid  = alloc((size_t)VTOT * 384 * 2);

  __bf16* xtok = (__bf16*)(ws + o_xtok);
  __bf16* wres = (__bf16*)(ws + o_wres);
  __bf16* wc1  = (__bf16*)(ws + o_wc1);
  __bf16* wc2  = (__bf16*)(ws + o_wc2);
  __bf16* wqkv = (__bf16*)(ws + o_wqkv);
  __bf16* wprj = (__bf16*)(ws + o_wprj);
  __bf16* wfc1 = (__bf16*)(ws + o_wfc1);
  __bf16* wfc2 = (__bf16*)(ws + o_wfc2);
  float* scr = (float*)(ws + o_scr); float* shr = (float*)(ws + o_shr);
  float* sc1 = (float*)(ws + o_sc1); float* sh1 = (float*)(ws + o_sh1);
  float* sc2 = (float*)(ws + o_sc2); float* sh2 = (float*)(ws + o_sh2);
  float*  cx  = (float*)(ws + o_cx);
  float*  t   = (float*)(ws + o_t);
  float*  res = (float*)(ws + o_res);
  __bf16* hbf = (__bf16*)(ws + o_hbf);
  __bf16* qkvb = (__bf16*)(ws + o_qkv);
  __bf16* aob = (__bf16*)(ws + o_aob);
  __bf16* hid = (__bf16*)(ws + o_hid);
  float*  out = (float*)d_out;

  // fold BN (+conv bias), pack activations & weights
  bn_fold_kernel<<<1, 96, 0, stream>>>(res_bn, res_b, scr, shr);
  bn_fold_kernel<<<1, 96, 0, stream>>>(bn1, conv1_b, sc1, sh1);
  bn_fold_kernel<<<1, 96, 0, stream>>>(bn2, conv2_b, sc2, sh2);
  pack_x_kernel<<<4096, 256, 0, stream>>>(x, xtok);
  pack_conv_frag_kernel<<<24, 256, 0, stream>>>(res_w, wres, CI, 2, 1);
  pack_conv_frag_kernel<<<648, 256, 0, stream>>>(conv1_w, wc1, CI, 2, 27);
  pack_conv_frag_kernel<<<972, 256, 0, stream>>>(conv2_w, wc2, CC, 3, 27);
  for (int dep = 0; dep < 2; ++dep) {
    pack_lin_frag_kernel<<<108, 256, 0, stream>>>(qkv_w + (size_t)dep * 288 * CC, wqkv + (size_t)dep * 3 * 18 * 512, 288, CC);
    pack_lin_frag_kernel<<<36, 256, 0, stream>>>(proj_w + (size_t)dep * CC * CC, wprj + (size_t)dep * 3 * 6 * 512, CC, CC);
    pack_lin_frag_kernel<<<144, 256, 0, stream>>>(fc1_w + (size_t)dep * 384 * CC, wfc1 + (size_t)dep * 3 * 24 * 512, 384, CC);
    pack_lin_frag_kernel<<<144, 256, 0, stream>>>(fc2_w + (size_t)dep * CC * 384, wfc2 + (size_t)dep * 12 * 6 * 512, CC, 384);
  }

  conv_res_kernel<<<MT / 4, 128, 0, stream>>>(xtok, wres, scr, shr, res);
  conv1_kernel<<<MT / 4, 128, 0, stream>>>(xtok, wc1, sc1, sh1, cx, t);

  const float qscale = 0.20412414523193154f;   // 24^-0.5
  for (int dep = 0; dep < 2; ++dep) {
    int shift = dep & 1;
    ln_kernel<<<VTOT / 256, 256, 0, stream>>>(t, n1 + (size_t)dep * 2 * CC, n1 + (size_t)dep * 2 * CC + CC, hbf);
    linear_kernel<<<(MT / 4) * 3, 128, 3 * 6144, stream>>>(
        hbf, wqkv + (size_t)dep * 3 * 18 * 512, qkv_b + (size_t)dep * 288, 288, CC,
        nullptr, qkvb, 0, qscale);
    attn_kernel<<<(16000 * NHD * 8) / 256, 256, 0, stream>>>(
        qkvb, rpb + (size_t)dep * 27 * NHD, aob, shift);
    linear_kernel<<<MT / 4, 128, 3 * 6144, stream>>>(
        aob, wprj + (size_t)dep * 3 * 6 * 512, proj_b + (size_t)dep * CC, CC, CC,
        t, nullptr, 1, 1.0f);
    ln_kernel<<<VTOT / 256, 256, 0, stream>>>(t, n2 + (size_t)dep * 2 * CC, n2 + (size_t)dep * 2 * CC + CC, hbf);
    linear_kernel<<<(MT / 4) * 4, 128, 3 * 6144, stream>>>(
        hbf, wfc1 + (size_t)dep * 3 * 24 * 512, fc1_b + (size_t)dep * 384, 384, CC,
        nullptr, hid, 2, 1.0f);
    linear_kernel<<<MT / 4, 128, 12 * 6144, stream>>>(
        hid, wfc2 + (size_t)dep * 12 * 6 * 512, fc2_b + (size_t)dep * CC, CC, 384,
        t, nullptr, 1, 1.0f);
  }

  cxsum_kernel<<<4096, 256, 0, stream>>>((const float4*)cx, (const float4*)t, (uint2*)hbf);
  conv2_kernel<<<MT / 4, 128, 0, stream>>>(hbf, wc2, sc2, sh2, res, out);
}
